// SelfAttentionLoss_5299989644011
// MI455X (gfx1250) — compile-verified, matrix operand order checked by reading, not run on hardware
//
#include <hip/hip_runtime.h>

#define N_LOSSES 8192
#define DIM 512
#define TQ 16
#define TK 64

typedef __bf16 bf16_t;
typedef __bf16 bf16x16 __attribute__((ext_vector_type(16)));
typedef __bf16 bf16x8  __attribute__((ext_vector_type(8)));
typedef float  f32x8   __attribute__((ext_vector_type(8)));

// Load one 16x32 bf16 tile (row stride ld) into the WMMA A/B-row fragment
// layout: lane = {idx 0..15, half 0..1}; dword v<4 -> K = 2v + 8*half,
// v>=4 -> K = 16 + 2(v-4) + 8*half. Consecutive K runs of 8 allow two
// 16-byte LDS loads (ds_load_b128) per lane.
__device__ __forceinline__ bf16x16 load_row_frag(const bf16_t* base, int ld, int lane) {
  const int m    = lane & 15;
  const int half = lane >> 4;
  const bf16_t* p = base + m * ld + half * 8;
  bf16x8 lo = *(const bf16x8*)(p);
  bf16x8 hi = *(const bf16x8*)(p + 16);
  bf16x16 f;
#pragma unroll
  for (int i = 0; i < 8; ++i) { f[i] = lo[i]; f[8 + i] = hi[i]; }
  return f;
}

// K index within a 32-wide chunk for fragment element i (see layout above).
__device__ __forceinline__ int kmap(int i, int half) {
  const int v = i >> 1;
  const int o = i & 1;
  return (v < 4) ? (2 * v + 8 * half + o) : (16 + 2 * (v - 4) + 8 * half + o);
}

// ---------------------------------------------------------------------------
// Kernel 1: collapse the rank-2 embedding through Wq/Wk/Wv.
//   aX = W_emb @ WX   (D-vector),  cX = b_emb @ WX + bX  (D-vector)
// so that  q[n,:] = l[n]*aq + cq, etc.
// ---------------------------------------------------------------------------
__global__ __launch_bounds__(DIM)
void prep_kernel(const float* __restrict__ W_emb, const float* __restrict__ b_emb,
                 const float* __restrict__ Wq, const float* __restrict__ bq,
                 const float* __restrict__ Wk, const float* __restrict__ bk,
                 const float* __restrict__ Wv, const float* __restrict__ bv,
                 float* __restrict__ vecs) {
  const int d = threadIdx.x;
  float aq = 0.f, cq = 0.f, ak = 0.f, ck = 0.f, av = 0.f, cv = 0.f;
  for (int e = 0; e < DIM; ++e) {
    const float we = W_emb[e], be = b_emb[e];
    const float wq = Wq[e * DIM + d]; aq = fmaf(we, wq, aq); cq = fmaf(be, wq, cq);
    const float wk = Wk[e * DIM + d]; ak = fmaf(we, wk, ak); ck = fmaf(be, wk, ck);
    const float wv = Wv[e * DIM + d]; av = fmaf(we, wv, av); cv = fmaf(be, wv, cv);
  }
  vecs[d]           = aq;
  vecs[DIM + d]     = cq + bq[d];
  vecs[2 * DIM + d] = ak;
  vecs[3 * DIM + d] = ck + bk[d];
  vecs[4 * DIM + d] = av;
  vecs[5 * DIM + d] = cv + bv[d];
}

// ---------------------------------------------------------------------------
// Kernel 2: flash attention, one 16-row query tile per block, 64-key tiles.
// 256 threads = 8 wave32 waves. Waves 0-3 produce the four 16x16 score tiles
// with v_wmma_f32_16x16x32_bf16 (16 k-steps each); softmax is parallel over
// all waves (2 rows/wave, shfl_xor tree in 16-lane groups); every wave owns a
// 16x64 output slice and folds in P@V (8 WMMAs) each iteration.
// ---------------------------------------------------------------------------
__global__ __launch_bounds__(256)
void flash_attn_kernel(const float* __restrict__ losses,
                       const float* __restrict__ vecs,
                       const float* __restrict__ W_out,
                       float* __restrict__ partials) {
  __shared__ bf16_t Qs[TQ * DIM];   // 16 KB
  __shared__ bf16_t Ks[TK * DIM];   // 64 KB
  __shared__ float  Ssm[TQ * TK];   // 4 KB (reused as reduction buffer)
  __shared__ bf16_t Psm[TQ * TK];   // 2 KB
  __shared__ float  ltile[TK];
  __shared__ float  rowm[TQ], rowl[TQ], rowc[TQ];

  const int tid  = threadIdx.x;
  const int wave = tid >> 5;
  const int lane = tid & 31;
  const int n    = lane & 15;
  const int half = lane >> 4;
  const int qt   = blockIdx.x;
  // (1/sqrt(512)) * log2(e): scores pre-scaled so softmax can use exp2.
  const float scale2 = 0.044194173824159216f * 1.4426950408889634f;

  const float* __restrict__ aq = vecs;
  const float* __restrict__ cq = vecs + DIM;
  const float* __restrict__ ak = vecs + 2 * DIM;
  const float* __restrict__ ck = vecs + 3 * DIM;
  const float* __restrict__ av = vecs + 4 * DIM;
  const float* __restrict__ cv = vecs + 5 * DIM;

  // Build the 16 x D query tile in bf16 (q = l*aq + cq).
  for (int e = tid; e < TQ * DIM; e += 256) {
    const int r = e >> 9;           // log2(DIM) = 9
    const int d = e & (DIM - 1);
    const float l = losses[qt * TQ + r];
    Qs[e] = (bf16_t)fmaf(l, aq[d], cq[d]);
  }
  if (tid < TQ) { rowm[tid] = -3.0e38f; rowl[tid] = 0.f; }

  // K columns this thread materializes each iteration.
  const int d0 = 2 * tid, d1 = 2 * tid + 1;
  const float akd0 = ak[d0], akd1 = ak[d1];
  const float ckd0 = ck[d0], ckd1 = ck[d1];

  // Per-lane V / W_out constants for this wave's 64 output columns.
  float avn[4], cvn[4], won[4];
#pragma unroll
  for (int ct = 0; ct < 4; ++ct) {
    const int dc = wave * 64 + ct * 16 + n;
    avn[ct] = av[dc]; cvn[ct] = cv[dc]; won[ct] = W_out[dc];
  }

  // Softmax row handled by this lane's 16-lane group.
  const int srow = 2 * wave + half;

  f32x8 acc[4] = {};  // O accumulator: 4 x (16x16 f32) = this wave's 64 cols

  for (int kt = 0; kt < N_LOSSES / TK; ++kt) {
    if (tid < TK) ltile[tid] = losses[kt * TK + tid];
    __builtin_prefetch(&losses[((kt + 1) * TK) & (N_LOSSES - 1)], 0, 0);
    __syncthreads();

    // Materialize K tile (bf16): k[j,d] = l[j]*ak[d] + ck[d].
#pragma unroll 4
    for (int j = 0; j < TK; ++j) {
      const float lj = ltile[j];
      Ks[j * DIM + d0] = (bf16_t)fmaf(lj, akd0, ckd0);
      Ks[j * DIM + d1] = (bf16_t)fmaf(lj, akd1, ckd1);
    }
    __syncthreads();

    // S = Q K^T * scale2 : four 16x16 tiles, K-dim 512 = 16 WMMA steps each.
    if (wave < 4) {
      f32x8 c = {};
#pragma unroll
      for (int ks = 0; ks < DIM / 32; ++ks) {
        bf16x16 a = load_row_frag(&Qs[ks * 32], DIM, lane);
        bf16x16 b = load_row_frag(&Ks[wave * 16 * DIM + ks * 32], DIM, lane);
        c = __builtin_amdgcn_wmma_f32_16x16x32_bf16(false, a, false, b,
                                                    (short)0, c, false, false);
      }
#pragma unroll
      for (int r = 0; r < 8; ++r)
        Ssm[(r + half * 8) * TK + wave * 16 + n] = c[r] * scale2;
    }
    __syncthreads();

    // Parallel online-softmax update: 2 rows per wave, 16 lanes per row,
    // 4 scores per lane; max/sum via shfl_xor tree within the lane group.
    {
      const float s0 = Ssm[srow * TK + n];
      const float s1 = Ssm[srow * TK + n + 16];
      const float s2 = Ssm[srow * TK + n + 32];
      const float s3 = Ssm[srow * TK + n + 48];
      float mx = fmaxf(fmaxf(s0, s1), fmaxf(s2, s3));
#pragma unroll
      for (int m = 1; m <= 8; m <<= 1) mx = fmaxf(mx, __shfl_xor(mx, m));
      const float mo   = rowm[srow];
      const float mnew = fmaxf(mo, mx);
      const float corr = exp2f(mo - mnew);
      const float p0 = exp2f(s0 - mnew);
      const float p1 = exp2f(s1 - mnew);
      const float p2 = exp2f(s2 - mnew);
      const float p3 = exp2f(s3 - mnew);
      Psm[srow * TK + n]      = (bf16_t)p0;
      Psm[srow * TK + n + 16] = (bf16_t)p1;
      Psm[srow * TK + n + 32] = (bf16_t)p2;
      Psm[srow * TK + n + 48] = (bf16_t)p3;
      float s = (p0 + p1) + (p2 + p3);
#pragma unroll
      for (int m = 1; m <= 8; m <<= 1) s += __shfl_xor(s, m);
      if (n == 0) {
        rowl[srow] = fmaf(rowl[srow], corr, s);
        rowm[srow] = mnew;
        rowc[srow] = corr;
      }
    }
    __syncthreads();

    // Rescale O, then O += P @ V (two k=32 chunks). V fragments synthesized
    // in registers from v[j,d] = l[j]*av[d] + cv[d] (column d fixed per lane).
    {
      float cr[8];
#pragma unroll
      for (int r = 0; r < 8; ++r) cr[r] = rowc[r + half * 8];
#pragma unroll
      for (int ct = 0; ct < 4; ++ct)
#pragma unroll
        for (int r = 0; r < 8; ++r) acc[ct][r] *= cr[r];

      const bf16x16 pf0 = load_row_frag(Psm, TK, lane);
      const bf16x16 pf1 = load_row_frag(Psm + 32, TK, lane);

      float ljv0[16], ljv1[16];
#pragma unroll
      for (int i = 0; i < 16; ++i) {
        const int k = kmap(i, half);
        ljv0[i] = ltile[k];
        ljv1[i] = ltile[32 + k];
      }
#pragma unroll
      for (int ct = 0; ct < 4; ++ct) {
        bf16x16 b0, b1;
#pragma unroll
        for (int i = 0; i < 16; ++i) {
          b0[i] = (bf16_t)fmaf(ljv0[i], avn[ct], cvn[ct]);
          b1[i] = (bf16_t)fmaf(ljv1[i], avn[ct], cvn[ct]);
        }
        acc[ct] = __builtin_amdgcn_wmma_f32_16x16x32_bf16(false, pf0, false, b0,
                                                          (short)0, acc[ct],
                                                          false, false);
        acc[ct] = __builtin_amdgcn_wmma_f32_16x16x32_bf16(false, pf1, false, b1,
                                                          (short)0, acc[ct],
                                                          false, false);
      }
    }
    __syncthreads();
  }

  // Epilogue: partial = sum over (rows, this wave's cols) of (O/l) * W_out.
  float local = 0.f;
  {
    float rl[8];
#pragma unroll
    for (int r = 0; r < 8; ++r) rl[r] = 1.f / rowl[r + half * 8];
#pragma unroll
    for (int ct = 0; ct < 4; ++ct)
#pragma unroll
      for (int r = 0; r < 8; ++r) local += acc[ct][r] * rl[r] * won[ct];
  }
  float* red = Ssm;  // >=256 floats, safe to reuse after last barrier
  red[tid] = local;
  __syncthreads();
  for (int s = 128; s > 0; s >>= 1) {
    if (tid < s) red[tid] += red[tid + s];
    __syncthreads();
  }
  if (tid == 0) partials[qt] = red[0];
}

// ---------------------------------------------------------------------------
// Kernel 3: reduce block partials -> mean + b_out.
// ---------------------------------------------------------------------------
__global__ __launch_bounds__(512)
void finalize_kernel(const float* __restrict__ partials,
                     const float* __restrict__ b_out,
                     float* __restrict__ out) {
  __shared__ float sm[512];
  const int t = threadIdx.x;
  sm[t] = partials[t];
  __syncthreads();
  for (int s = 256; s > 0; s >>= 1) {
    if (t < s) sm[t] += sm[t + s];
    __syncthreads();
  }
  if (t == 0) out[0] = sm[0] * (1.0f / N_LOSSES) + b_out[0];
}

extern "C" void kernel_launch(void* const* d_in, const int* in_sizes, int n_in,
                              void* d_out, int out_size, void* d_ws, size_t ws_size,
                              hipStream_t stream) {
  const float* losses = (const float*)d_in[0];
  const float* W_emb  = (const float*)d_in[1];
  const float* b_emb  = (const float*)d_in[2];
  const float* Wq     = (const float*)d_in[3];
  const float* bq     = (const float*)d_in[4];
  const float* Wk     = (const float*)d_in[5];
  const float* bk     = (const float*)d_in[6];
  const float* Wv     = (const float*)d_in[7];
  const float* bv     = (const float*)d_in[8];
  const float* W_out  = (const float*)d_in[9];
  const float* b_out  = (const float*)d_in[10];
  float* out = (float*)d_out;

  float* vecs     = (float*)d_ws;      // 6*DIM floats: aq,cq,ak,ck,av,cv
  float* partials = vecs + 6 * DIM;    // N/TQ = 512 floats

  prep_kernel<<<1, DIM, 0, stream>>>(W_emb, b_emb, Wq, bq, Wk, bk, Wv, bv, vecs);
  flash_attn_kernel<<<N_LOSSES / TQ, 256, 0, stream>>>(losses, vecs, W_out, partials);
  finalize_kernel<<<1, 512, 0, stream>>>(partials, b_out, out);
}
